// CausalFlexAttention_66623532696048
// MI455X (gfx1250) — compile-verified, matrix operand order checked by reading, not run on hardware
//
#include <hip/hip_runtime.h>
#include <hip/hip_bf16.h>

typedef _Float16 v16h __attribute__((ext_vector_type(16)));
typedef float    v8f  __attribute__((ext_vector_type(8)));

// Problem constants (B,V,L,C,H) = (2,8,256,1024,16)
constexpr int B0 = 2;
constexpr int H0 = 16;
constexpr int DH = 64;
constexpr int C0 = 1024;    // channels = K-dim of projections
constexpr int S0 = 2048;    // V*L sequence length per batch
constexpr int M0 = B0 * S0; // 4096 GEMM rows

// ---- CDNA5 async copy helpers (ASYNCcnt-tracked global->LDS DMA) ----------
__device__ __forceinline__ unsigned lds_off(const void* p) {
    // generic shared pointer: low 32 bits == LDS byte offset
    return (unsigned)(unsigned long long)p;
}
__device__ __forceinline__ void async_b128(const void* gaddr, unsigned ldst) {
    asm volatile("global_load_async_to_lds_b128 %0, %1, off"
                 :: "v"(ldst), "v"(gaddr) : "memory");
}
__device__ __forceinline__ void wait_async0() {
    asm volatile("s_wait_asynccnt 0" ::: "memory");
}

// ---- DPP16 butterfly max over each 16-lane half (no LDS, no index math) ---
template<int CTRL>
__device__ __forceinline__ float fmax_dpp(float x) {
    const int t = __builtin_amdgcn_update_dpp(0, __float_as_int(x),
                                              CTRL, 0xF, 0xF, true);
    return fmaxf(x, __int_as_float(t));
}
__device__ __forceinline__ float rowmax16(float x) {
    x = fmax_dpp<0xB1>(x);   // quad_perm(1,0,3,2) : xor 1
    x = fmax_dpp<0x4E>(x);   // quad_perm(2,3,0,1) : xor 2
    x = fmax_dpp<0x141>(x);  // row_half_mirror    : fold quads in 8-group
    x = fmax_dpp<0x140>(x);  // row_mirror         : fold 8-groups in row
    return x;
}

// ---------------------------------------------------------------------------
// One-shot f32 -> f16 conversion (x and the four weight matrices), so all
// GEMM/attention staging becomes pure async b128 copies with no VALU convert.
// ---------------------------------------------------------------------------
__global__ __launch_bounds__(256)
void cvt_f32_f16(const float* __restrict__ in, _Float16* __restrict__ out) {
    const int i = (blockIdx.x * 256 + threadIdx.x) * 8;
    #pragma unroll
    for (int j = 0; j < 8; ++j) out[i + j] = (_Float16)in[i + j];
}

// ---------------------------------------------------------------------------
// Fused WMMA GEMM, double-buffered LDS, all staging via async-to-LDS:
//   MODE 0: A = x (f16), 3 weights -> Q (f16 row), K (f16 row), V^T (f16 B,C,S)
//   MODE 1: A = AO (f16), 1 weight -> out (f32 row) + bias
// Block tile 128x64, 8 waves, each wave 32x32 via 2x2 WMMA tiles.
// WMMA issue is grouped by A-fragment (operand-reuse friendly).
// ---------------------------------------------------------------------------
constexpr int TM = 128, TN = 64, TK = 32;
constexpr int LDT = TK + 16;  // 48 halves = 96B rows: v16h reads stay 32B-aligned

template<int MODE>
__global__ __launch_bounds__(256)
void gemm_fused(const _Float16* __restrict__ A,
                const _Float16* __restrict__ W0h, const _Float16* __restrict__ W1h,
                const _Float16* __restrict__ W2h, const float* __restrict__ bias,
                _Float16* __restrict__ Qh, _Float16* __restrict__ Kh,
                _Float16* __restrict__ Vt, float* __restrict__ outF)
{
    constexpr int NW = (MODE == 0) ? 3 : 1;
    __shared__ __align__(32) _Float16 la[2][TM * LDT];
    __shared__ __align__(32) _Float16 lb[2][NW][TN * LDT];

    const int tid  = threadIdx.x;
    const int lane = tid & 31;
    const int wave = tid >> 5;
    const int wy   = wave >> 1;    // 0..3 : 32-row band
    const int wx   = wave & 1;     // 0..1 : 32-col band
    const int m0   = blockIdx.x * TM;
    const int n0   = blockIdx.y * TN;

    const int rl    = lane & 15;
    const int khalf = lane >> 4;

    const _Float16* Wh[3] = { W0h, W1h, W2h };

    auto stage = [&](int k0, int buf) {
        // A tile 128x32: 2 threads/row, 32B each -> 2 async b128
        const int r  = tid >> 1;
        const int kh = (tid & 1) * 16;
        const _Float16* asrc = A + (size_t)(m0 + r) * C0 + k0 + kh;
        const unsigned  adst = lds_off(&la[buf][r * LDT + kh]);
        async_b128(asrc, adst);
        async_b128(asrc + 8, adst + 16);
        // Weight tiles 64x32: 4 threads/row, 16B each -> 1 async b128 per weight
        const int rw = tid >> 2;
        const int kw = (tid & 3) * 8;
        #pragma unroll
        for (int w = 0; w < NW; ++w) {
            const _Float16* wsrc = Wh[w] + (size_t)(n0 + rw) * C0 + k0 + kw;
            async_b128(wsrc, lds_off(&lb[buf][w][rw * LDT + kw]));
        }
    };

    v8f acc[NW][2][2] = {};

    stage(0, 0);
    wait_async0();
    __syncthreads();

    for (int k0 = 0, it = 0; k0 < C0; k0 += TK, ++it) {
        const int cur = it & 1;
        if (k0 + TK < C0) stage(k0 + TK, cur ^ 1);   // async DMA overlaps WMMA below

        v16h af[2];
        #pragma unroll
        for (int s = 0; s < 2; ++s)
            af[s] = *(const v16h*)&la[cur][(wy * 32 + s * 16 + rl) * LDT + khalf * 16];

        v16h bf[NW][2];
        #pragma unroll
        for (int w = 0; w < NW; ++w)
            #pragma unroll
            for (int s = 0; s < 2; ++s)
                bf[w][s] = *(const v16h*)&lb[cur][w][(wx * 32 + s * 16 + rl) * LDT + khalf * 16];

        // group by A-fragment: af[i] constant across 2*NW consecutive WMMAs
        #pragma unroll
        for (int i = 0; i < 2; ++i)
            #pragma unroll
            for (int w = 0; w < NW; ++w)
                #pragma unroll
                for (int j = 0; j < 2; ++j)
                    acc[w][i][j] = __builtin_amdgcn_wmma_f32_16x16x32_f16(
                        false, af[i], false, bf[w][j], (short)0, acc[w][i][j], false, false);
        wait_async0();
        __syncthreads();
    }

    // ---- writeback: D row = e (lanes 0-15) / e+8 (lanes 16-31), col = lane&15
    const int rhi = khalf << 3;
    #pragma unroll
    for (int w = 0; w < NW; ++w) {
        #pragma unroll
        for (int i = 0; i < 2; ++i) {
            #pragma unroll
            for (int j = 0; j < 2; ++j) {
                const int mbase = m0 + wy * 32 + i * 16 + rhi;
                const int nbase = n0 + wx * 32 + j * 16 + rl;
                #pragma unroll
                for (int e = 0; e < 8; ++e) {
                    const int m = mbase + e;
                    const float v = acc[w][i][j][e];
                    if constexpr (MODE == 0) {
                        if (w == 0)      Qh[(size_t)m * C0 + nbase] = (_Float16)v;
                        else if (w == 1) Kh[(size_t)m * C0 + nbase] = (_Float16)v;
                        else {
                            const int b = m >> 11;           // / S0
                            const int s = m & (S0 - 1);
                            Vt[((size_t)b * C0 + nbase) * S0 + s] = (_Float16)v;
                        }
                    } else {
                        outF[(size_t)m * C0 + nbase] = v + bias[nbase];
                    }
                }
            }
        }
    }
}

// ---------------------------------------------------------------------------
// Block-sparse flash attention with cooperative async K/V staging.
// View mask is constant over 256x256 blocks -> skip disallowed blocks.
// All 8 waves share (b,h) and the key-slab stream, so each 64-key K slab
// (64x64 f16, 8KB) and V^T slab (8KB) is DMA'd into LDS once per block
// (double-buffered), then every wave reads fragments from LDS.
// Softmax runs in base-2 with dh^-0.5*log2(e) pre-folded into the Q
// fragments; row-max uses a 4-step DPP16 butterfly (no LDS shuffles);
// row sums ride the matrix pipe (P x ones).
// ---------------------------------------------------------------------------
__global__ __launch_bounds__(256)
void attn_flash(const _Float16* __restrict__ Q, const _Float16* __restrict__ Kh,
                const _Float16* __restrict__ Vt, _Float16* __restrict__ AO)
{
    __shared__ __align__(32) _Float16 lk[2][64 * 64];     // keys x d
    __shared__ __align__(32) _Float16 lv[2][64 * 64];     // d x keys
    __shared__ __align__(32) _Float16 pbuf[8 * 16 * 64];  // per-wave 16x64 probs

    const int tid   = threadIdx.x;
    const int lane  = tid & 31;
    const int wave  = tid >> 5;
    const int rl    = lane & 15;
    const int khalf = lane >> 4;

    const int q0 = blockIdx.x * 128 + wave * 16;
    const int bh = blockIdx.y;
    const int b  = bh >> 4;          // / H0
    const int h  = bh & (H0 - 1);
    const int qv = q0 >> 8;          // view index (L = 256), block-uniform

    // flattened, block-uniform list of allowed 256-key views (4 slabs each)
    int kvlist[8]; int nkv = 0;
    #pragma unroll
    for (int kv = 0; kv < 8; ++kv) {
        const bool allow = ((qv == 0) && (kv == 1)) || ((qv == 1) && (kv == 0)) ||
                           (((qv > 1) || (kv > 1)) && (qv >= kv));
        if (allow) kvlist[nkv++] = kv;
    }
    const int nslab = nkv * 4;

    auto stageKV = [&](int idx, int buf) {
        const int k0  = kvlist[idx >> 2] * 256 + (idx & 3) * 64;
        const int row = tid >> 2;          // 0..63
        const int ch  = (tid & 3) * 16;    // halves
        const _Float16* ks = Kh + (size_t)(b * S0 + k0 + row) * C0 + h * DH + ch;
        const unsigned  kd = lds_off(&lk[buf][row * 64 + ch]);
        async_b128(ks, kd);
        async_b128(ks + 8, kd + 16);
        const _Float16* vs = Vt + ((size_t)b * C0 + h * DH + row) * S0 + k0 + ch;
        const unsigned  vd = lds_off(&lv[buf][row * 64 + ch]);
        async_b128(vs, vd);
        async_b128(vs + 8, vd + 16);
    };

    _Float16* pw = pbuf + wave * (16 * 64);

    v16h ones;
    #pragma unroll
    for (int i = 0; i < 16; ++i) ones[i] = (_Float16)1.0f;

    // Q fragments: 16 queries x 64 d, pre-scaled by dh^-0.5 * log2(e)
    const _Float16 qs = (_Float16)(0.125f * 1.44269504088896f);
    const _Float16* qp = Q + ((size_t)(b * S0 + q0 + rl)) * C0 + h * DH + khalf * 16;
    v16h qa[2];
    qa[0] = *(const v16h*)qp;
    qa[1] = *(const v16h*)(qp + 32);
    #pragma unroll
    for (int i = 0; i < 16; ++i) { qa[0][i] *= qs; qa[1][i] *= qs; }

    v8f   acc[4] = {};
    float mrun[8], lrun[8];
    #pragma unroll
    for (int e = 0; e < 8; ++e) { mrun[e] = -3.0e30f; lrun[e] = 0.0f; }

    stageKV(0, 0);
    wait_async0();
    __syncthreads();

    for (int idx = 0; idx < nslab; ++idx) {
        const int cur = idx & 1;
        if (idx + 1 < nslab) stageKV(idx + 1, cur ^ 1);   // DMA overlaps compute

        // ---- scores (log2-domain): grouped so the A operand stays constant ----
        v16h kb0[4], kb1[4];
        #pragma unroll
        for (int j = 0; j < 4; ++j) {
            const _Float16* kp = &lk[cur][(j * 16 + rl) * 64 + khalf * 16];
            kb0[j] = *(const v16h*)kp;
            kb1[j] = *(const v16h*)(kp + 32);
        }
        v8f s[4];
        #pragma unroll
        for (int j = 0; j < 4; ++j)
            s[j] = __builtin_amdgcn_wmma_f32_16x16x32_f16(false, qa[0], false, kb0[j], (short)0, v8f{}, false, false);
        #pragma unroll
        for (int j = 0; j < 4; ++j)
            s[j] = __builtin_amdgcn_wmma_f32_16x16x32_f16(false, qa[1], false, kb1[j], (short)0, s[j], false, false);

        // ---- online softmax (base 2): DPP row-max, exp2, stage P ----
        float alph[8];
        #pragma unroll
        for (int e = 0; e < 8; ++e) {
            float mx = fmaxf(fmaxf(s[0][e], s[1][e]), fmaxf(s[2][e], s[3][e]));
            mx = rowmax16(mx);
            const float mnew = fmaxf(mrun[e], mx);
            alph[e] = exp2f(mrun[e] - mnew);
            mrun[e] = mnew;
            const float e0 = exp2f(s[0][e] - mnew);
            const float e1 = exp2f(s[1][e] - mnew);
            const float e2 = exp2f(s[2][e] - mnew);
            const float e3 = exp2f(s[3][e] - mnew);
            #pragma unroll
            for (int t = 0; t < 4; ++t) acc[t][e] *= alph[e];
            const int prow = e + (khalf << 3);
            pw[prow * 64 + rl]      = (_Float16)e0;
            pw[prow * 64 + 16 + rl] = (_Float16)e1;
            pw[prow * 64 + 32 + rl] = (_Float16)e2;
            pw[prow * 64 + 48 + rl] = (_Float16)e3;
        }

        // ---- P fragments (two K=32 slabs) ----
        const v16h pf0 = *(const v16h*)&pw[rl * 64 + khalf * 16];
        const v16h pf1 = *(const v16h*)&pw[rl * 64 + 32 + khalf * 16];

        // ---- row sums on the matrix pipe: rsum = P x ones(64x16) ----
        v8f rsum;
        rsum = __builtin_amdgcn_wmma_f32_16x16x32_f16(false, pf0, false, ones, (short)0, v8f{}, false, false);
        rsum = __builtin_amdgcn_wmma_f32_16x16x32_f16(false, pf1, false, ones, (short)0, rsum, false, false);
        #pragma unroll
        for (int e = 0; e < 8; ++e) lrun[e] = lrun[e] * alph[e] + rsum[e];

        // ---- P.V, grouped by P fragment: 4 d-tiles x 2 key slabs ----
        v16h vf0[4], vf1[4];
        #pragma unroll
        for (int t = 0; t < 4; ++t) {
            const _Float16* vp = &lv[cur][(t * 16 + rl) * 64 + khalf * 16];
            vf0[t] = *(const v16h*)vp;
            vf1[t] = *(const v16h*)(vp + 32);
        }
        #pragma unroll
        for (int t = 0; t < 4; ++t)
            acc[t] = __builtin_amdgcn_wmma_f32_16x16x32_f16(false, pf0, false, vf0[t], (short)0, acc[t], false, false);
        #pragma unroll
        for (int t = 0; t < 4; ++t)
            acc[t] = __builtin_amdgcn_wmma_f32_16x16x32_f16(false, pf1, false, vf1[t], (short)0, acc[t], false, false);

        wait_async0();     // next slab's DMA landed
        __syncthreads();   // all waves done with buf cur, buf cur^1 ready
    }

    // ---- normalize and store attention output f16 (B,S,C) ----
    #pragma unroll
    for (int e = 0; e < 8; ++e) {
        const float inv = 1.0f / lrun[e];
        const int   m   = b * S0 + q0 + e + (khalf << 3);
        #pragma unroll
        for (int t = 0; t < 4; ++t) {
            AO[(size_t)m * C0 + h * DH + t * 16 + rl] = (_Float16)(acc[t][e] * inv);
        }
    }
}

// ---------------------------------------------------------------------------
extern "C" void kernel_launch(void* const* d_in, const int* in_sizes, int n_in,
                              void* d_out, int out_size, void* d_ws, size_t ws_size,
                              hipStream_t stream)
{
    (void)in_sizes; (void)n_in; (void)out_size; (void)ws_size;
    const float* x  = (const float*)d_in[0];
    const float* Wq = (const float*)d_in[1];
    const float* Wk = (const float*)d_in[2];
    const float* Wv = (const float*)d_in[3];
    const float* Wp = (const float*)d_in[4];
    const float* bp = (const float*)d_in[5];
    float* out = (float*)d_out;

    const size_t N  = (size_t)M0 * C0;     // 4096*1024
    const size_t WN = (size_t)C0 * C0;     // 1024*1024
    _Float16* Qh  = (_Float16*)d_ws;       // (B,S,C) f16
    _Float16* Kh  = Qh + N;                // (B,S,C) f16
    _Float16* Vt  = Kh + N;                // (B,C,S) f16 (transposed)
    _Float16* AO  = Vt + N;                // (B,S,C) f16 attention output
    _Float16* xh  = AO + N;                // x in f16
    _Float16* wqh = xh + N;
    _Float16* wkh = wqh + WN;
    _Float16* wvh = wkh + WN;
    _Float16* wph = wvh + WN;

    // one-shot f32 -> f16 rewrites (pure bandwidth, enables async-only staging)
    cvt_f32_f16<<<(int)(N  / (8 * 256)), 256, 0, stream>>>(x,  xh);
    cvt_f32_f16<<<(int)(WN / (8 * 256)), 256, 0, stream>>>(Wq, wqh);
    cvt_f32_f16<<<(int)(WN / (8 * 256)), 256, 0, stream>>>(Wk, wkh);
    cvt_f32_f16<<<(int)(WN / (8 * 256)), 256, 0, stream>>>(Wv, wvh);
    cvt_f32_f16<<<(int)(WN / (8 * 256)), 256, 0, stream>>>(Wp, wph);

    dim3 gg(M0 / TM, C0 / TN);             // 32 x 16 workgroups, 256 threads

    gemm_fused<0><<<gg, 256, 0, stream>>>(xh, wqh, wkh, wvh, nullptr,
                                          Qh, Kh, Vt, nullptr);

    attn_flash<<<dim3(S0 / 128, B0 * H0), 256, 0, stream>>>(Qh, Kh, Vt, AO);

    gemm_fused<1><<<gg, 256, 0, stream>>>(AO, wph, nullptr, nullptr, bp,
                                          nullptr, nullptr, nullptr, out);
}
